// HypergraphConv_36790689857779
// MI455X (gfx1250) — compile-verified, matrix operand order checked by reading, not run on hardware
//
#include <hip/hip_runtime.h>
#include <hip/hip_bf16.h>

#define EPS_F 1e-6f

typedef __attribute__((ext_vector_type(2)))  float  v2f;
typedef __attribute__((ext_vector_type(8)))  float  v8f;
typedef __attribute__((ext_vector_type(16))) __bf16 v16bf;

#define HAVE_WMMA_F32X4 __has_builtin(__builtin_amdgcn_wmma_f32_16x16x4_f32)

// ---------------------------------------------------------------------------
// Kernel 1: zero the output accumulator, init per-node degree d[] = EPS
// ---------------------------------------------------------------------------
__global__ void hgc_init(float* __restrict__ acc, float* __restrict__ d,
                         int N, int M) {
    int i = blockIdx.x * blockDim.x + threadIdx.x;
    if (i < N * M) acc[i] = 0.0f;
    if (i < N)     d[i]   = EPS_F;
}

// ---------------------------------------------------------------------------
// Kernel 2: per-edge prep. Only 2 gathered values of H_hat_full matter per
// column. b_e is local; d[] needs atomic scatter-add.
// ED[e] = {hs, ht, hs*w/b, ht*w/b}
// ---------------------------------------------------------------------------
__global__ void hgc_edge_prep(const int* __restrict__ EI,
                              const float* __restrict__ H,
                              const float* __restrict__ EW,
                              float* __restrict__ d,
                              float4* __restrict__ ED,
                              int E) {
    int e = blockIdx.x * blockDim.x + threadIdx.x;
    if (e >= E) return;
    int s = EI[e];
    int t = EI[E + e];
    float hs = fmaxf(H[(size_t)s * E + e], 0.0f);
    float ht = (s == t) ? 0.0f : fmaxf(H[(size_t)t * E + e], 0.0f);
    float w  = fmaxf(EW[e], 0.0f);
    float binv = w / (hs + ht + EPS_F);           // w_e / b_e
    ED[e] = make_float4(hs, ht, hs * binv, ht * binv);
    atomicAdd(&d[s], hs * w);
    if (s != t) atomicAdd(&d[t], ht * w);
}

// ---------------------------------------------------------------------------
// Kernel 3: x_lin = X @ W^T + bias via WMMA. One wave per 16x16 tile.
// fp32 path: V_WMMA_F32_16X16X4_F32 (exact fp32 like the reference).
// A 16x4 layout: lanes 0-15 -> K+{0,1}, lanes 16-31 -> K+{2,3} (2 VGPRs).
// B 4x16 mirrors (lane&15 = output column = row of W).
// C/D: VGPR r holds row r (lanes 0-15) / r+8 (lanes 16-31), col = lane&15.
// ---------------------------------------------------------------------------
__global__ __launch_bounds__(256)
void hgc_gemm_wmma(const float* __restrict__ X, const float* __restrict__ W,
                   const float* __restrict__ B, float* __restrict__ XL,
                   int N, int K, int M) {
    const int lane   = threadIdx.x & 31;
    const int wave   = threadIdx.x >> 5;
    const int tilesO = M >> 4;
    const int tile   = blockIdx.x * 8 + wave;
    const int n0     = (tile / tilesO) << 4;
    const int o0     = (tile % tilesO) << 4;
    if (n0 >= N) return;                 // wave-uniform: EXEC stays all-ones

    const int l15 = lane & 15;
    v8f acc = {0.f, 0.f, 0.f, 0.f, 0.f, 0.f, 0.f, 0.f};

#if HAVE_WMMA_F32X4
    const int koff = (lane >> 4) << 1;   // 0 or 2
    const float* xrow = X + (size_t)(n0 + l15) * K + koff;
    const float* wrow = W + (size_t)(o0 + l15) * K + koff;
    __builtin_prefetch(xrow, 0, 3);
    __builtin_prefetch(wrow, 0, 3);
#pragma unroll 8
    for (int k = 0; k < K; k += 4) {
        v2f a = *(const v2f*)(xrow + k);
        v2f b = *(const v2f*)(wrow + k);
        acc = __builtin_amdgcn_wmma_f32_16x16x4_f32(
            false, a, false, b, (short)0, acc, false, false);
    }
#else
    // Fallback: bf16 WMMA with on-the-fly convert (codegen-confirmed builtin).
    // 16-bit A 16x32: lanes0-15 VGPR0-3 = K 0..7, VGPR4-7 = K 16..23;
    // lanes16-31 shifted by +8. B assumed symmetric with lane&15 = column.
    const int h8 = (lane >> 4) << 3;     // 0 or 8
    const float* xrow = X + (size_t)(n0 + l15) * K;
    const float* wrow = W + (size_t)(o0 + l15) * K;
    for (int k = 0; k < K; k += 32) {
        v16bf a, b;
#pragma unroll
        for (int j = 0; j < 8; ++j) {
            a[j]     = (__bf16)xrow[k + h8 + j];
            a[j + 8] = (__bf16)xrow[k + 16 + h8 + j];
            b[j]     = (__bf16)wrow[k + h8 + j];
            b[j + 8] = (__bf16)wrow[k + 16 + h8 + j];
        }
        acc = __builtin_amdgcn_wmma_f32_16x16x32_bf16(
            false, a, false, b, (short)0, acc, false, false);
    }
#endif

    const float bv    = B[o0 + l15];
    const int   colg  = o0 + l15;
    const int   rbase = n0 + ((lane >> 4) << 3);
#pragma unroll
    for (int r = 0; r < 8; ++r)
        XL[(size_t)(rbase + r) * M + colg] = acc[r] + bv;
}

// ---------------------------------------------------------------------------
// Kernel 4: per-edge fused tmp + scatter. One block (M threads) per edge.
// v = hs*x_lin[s] + ht*x_lin[t]  (== tmp[e]);  acc[s] += cs*v; acc[t] += ct*v
// ---------------------------------------------------------------------------
__global__ __launch_bounds__(256)
void hgc_scatter(const int* __restrict__ EI, const float4* __restrict__ ED,
                 const float* __restrict__ XL, float* __restrict__ acc,
                 int E, int M) {
    int e = blockIdx.x;
    int j = threadIdx.x;
    int s = EI[e];
    int t = EI[E + e];
    float4 ed = ED[e];
    float v = ed.x * XL[(size_t)s * M + j];
    if (s != t) v += ed.y * XL[(size_t)t * M + j];
    atomicAdd(&acc[(size_t)s * M + j], ed.z * v);
    if (s != t) atomicAdd(&acc[(size_t)t * M + j], ed.w * v);
}

// ---------------------------------------------------------------------------
// Kernel 5: out = relu(acc / d[n]) in place
// ---------------------------------------------------------------------------
__global__ void hgc_finalize(float* __restrict__ acc, const float* __restrict__ d,
                             int N, int M) {
    int i = blockIdx.x * blockDim.x + threadIdx.x;
    if (i >= N * M) return;
    float r = acc[i] / d[i / M];
    acc[i] = fmaxf(r, 0.0f);
}

// ---------------------------------------------------------------------------
extern "C" void kernel_launch(void* const* d_in, const int* in_sizes, int n_in,
                              void* d_out, int out_size, void* d_ws, size_t ws_size,
                              hipStream_t stream) {
    const float* x  = (const float*)d_in[0];
    const int*   ei = (const int*)d_in[1];
    const float* hh = (const float*)d_in[2];
    const float* ew = (const float*)d_in[3];
    const float* wl = (const float*)d_in[4];
    const float* bl = (const float*)d_in[5];
    float* out = (float*)d_out;

    const int E = in_sizes[1] / 2;               // 4096
    const int M = in_sizes[5];                   // d_out = 256
    const int K = in_sizes[4] / M;               // d_in  = 256
    const int N = in_sizes[0] / K;               // 4096

    // Workspace layout: x_lin (N*M f32) | d (N f32) | edge data (E float4)
    char*   ws = (char*)d_ws;
    float*  xl = (float*)ws;
    float*  dd = (float*)(ws + (size_t)N * M * sizeof(float));
    float4* ed = (float4*)(ws + (size_t)N * M * sizeof(float)
                              + (size_t)N * sizeof(float));

    const int total = N * M;
    hgc_init<<<(total + 255) / 256, 256, 0, stream>>>(out, dd, N, M);
    hgc_edge_prep<<<(E + 255) / 256, 256, 0, stream>>>(ei, hh, ew, dd, ed, E);
    const int tiles = (N >> 4) * (M >> 4);       // one wave each, 8 waves/block
    hgc_gemm_wmma<<<(tiles + 7) / 8, 256, 0, stream>>>(x, wl, bl, xl, N, K, M);
    hgc_scatter<<<E, M, 0, stream>>>(ei, ed, xl, out, E, M);
    hgc_finalize<<<(total + 255) / 256, 256, 0, stream>>>(out, dd, N, M);
}
